// unaryNet_6390911336641
// MI455X (gfx1250) — matrix operand order repaired and checked by reading, not compiled, not run on hardware
//
#include <hip/hip_runtime.h>
#include <hip/hip_bf16.h>
#include <stdint.h>

// ---------------------------------------------------------------------------
// Unary-computing MLP (1024->512->256->10), T=256 bit-stream steps, B=16.
// th[t,b,i] thresholds / s[t,b,i] signs are prefix-sum derived (parallel over
// t); heavy compare-count kernels build f16 0/1 A-fragments with VALU
// (v_cmp_gt_f32 + v_cndmask_b16) and absorb the reduction into
// V_WMMA_F32_16X16X32_F16.  th stored f32 (no per-use cvt in the hot loop),
// s stored f16 (feeds B-fragment directly).
// ---------------------------------------------------------------------------

#define T_STEPS 256
#define BATCH   16
#define IN1     1024
#define H1      512
#define H2      256
#define NOUT    10

typedef __attribute__((ext_vector_type(16))) _Float16 v16h;
typedef __attribute__((ext_vector_type(8)))  float    v8f;

// Detect gfx1250 async global->LDS builtins (guarded; fallback = plain copy).
#if defined(__has_builtin)
#if __has_builtin(__builtin_amdgcn_global_load_async_to_lds_b32)
#define HAVE_ASYNC_LDS 1
#endif
#endif
#ifndef HAVE_ASYNC_LDS
#define HAVE_ASYNC_LDS 0
#endif

typedef __attribute__((address_space(1))) int* gptr_i32;
typedef __attribute__((address_space(3))) int* lptr_i32;

// 1-D unscrambled Sobol value k (k in [0,255]) mapped to [-1,1):
// rng(k) = bitreverse8(gray(k))/128 - 1.
__device__ __forceinline__ float sobol_val(int k) {
  unsigned g = (unsigned)k ^ ((unsigned)k >> 1);
  g = __brev(g) >> 24; // bitreverse8
  return (float)g * (1.0f / 128.0f) - 1.0f;
}

// ---------------------------------------------------------------------------
// K1: layer-1 prep. One block per batch element b, 1024 threads (i).
// Produces th1[t,b,i] (f32), s1[t,b,i] = +/-1 (f16),
// cnt0_1[t,b] = sum_i (1 - x[t,b,i]).
// ---------------------------------------------------------------------------
__global__ void prep1_kernel(const int* __restrict__ x,
                             float* __restrict__ th1,
                             _Float16* __restrict__ s1,
                             float* __restrict__ cnt0) {
  __shared__ float rng[T_STEPS];
  __shared__ int   cnt[T_STEPS];
  const int b = blockIdx.x;
  const int i = threadIdx.x; // 0..1023
  if (i < T_STEPS) { rng[i] = sobol_val(i); cnt[i] = 0; }
  __syncthreads();

  int cum = 0;
  for (int t = 0; t < T_STEPS; ++t) {
    const size_t idx = ((size_t)t * BATCH + b) * IN1 + i;
    const int xi = x[idx];
    th1[idx] = xi ? rng[cum] : rng[t - cum]; // cum_inv = t - cum
    s1[idx]  = xi ? (_Float16)1.0f : (_Float16)(-1.0f);
    int na = 1 - xi;
#pragma unroll
    for (int d = 16; d; d >>= 1) na += __shfl_down(na, d, 32);
    if ((i & 31) == 0) atomicAdd(&cnt[t], na); // ds_add_u32
    cum += xi;
  }
  __syncthreads();
  if (i < T_STEPS) cnt0[(size_t)i * BATCH + b] = (float)cnt[i];
}

// ---------------------------------------------------------------------------
// K2: heavy compare-count kernel using WMMA.
// Grid: (T, B, FOUT/128); block = 256 threads = 8 waves, each wave owns a
// 16-row output tile. Per 32-wide K-chunk, VALU builds the 16x32 f16
// A-fragment of compare bits (w[o,i] > th[t,b,i]) per the CDNA5 16-bit
// A-matrix layout; B holds s[t,b,i] (+/-1) replicated to all 16 columns, so
// every D column accumulates the same y = sum_i s_i * gt_i.
// ---------------------------------------------------------------------------
template <int FIN, int FOUT>
__global__ void ulinear_wmma_kernel(const float* __restrict__ W,       // [FOUT, FIN]
                                    const float* __restrict__ thG,     // [T,B,FIN] f32
                                    const _Float16* __restrict__ sG,   // [T,B,FIN] f16
                                    float* __restrict__ yraw) {        // [T,B,FOUT]
  static_assert(FIN % 32 == 0 && FOUT % 128 == 0, "tile shapes");
  __shared__ float    thS[FIN];
  __shared__ _Float16 sS[FIN];

  const int t = blockIdx.x, b = blockIdx.y;
  const size_t tb = (size_t)t * BATCH + b;

  // Stage th/s tile (shared by all 8 waves) into LDS, async when available.
  const uint32_t* thg = reinterpret_cast<const uint32_t*>(thG + tb * FIN);
  const uint32_t* sg  = reinterpret_cast<const uint32_t*>(sG + tb * FIN);
  uint32_t* thl = reinterpret_cast<uint32_t*>(thS);
  uint32_t* sl  = reinterpret_cast<uint32_t*>(sS);
#if HAVE_ASYNC_LDS
  for (int k = threadIdx.x; k < FIN; k += 256) {
    __builtin_amdgcn_global_load_async_to_lds_b32(
        (gptr_i32)(uintptr_t)(thg + k), (lptr_i32)(uintptr_t)(thl + k), 0, 0);
  }
  for (int k = threadIdx.x; k < FIN / 2; k += 256) {
    __builtin_amdgcn_global_load_async_to_lds_b32(
        (gptr_i32)(uintptr_t)(sg + k), (lptr_i32)(uintptr_t)(sl + k), 0, 0);
  }
#if __has_builtin(__builtin_amdgcn_s_wait_asynccnt)
  __builtin_amdgcn_s_wait_asynccnt(0);
#else
  asm volatile("s_wait_asynccnt 0" ::: "memory");
#endif
#else
  for (int k = threadIdx.x; k < FIN; k += 256) thl[k] = thg[k];
  for (int k = threadIdx.x; k < FIN / 2; k += 256) sl[k] = sg[k];
#endif
  __syncthreads();

  const int lane = threadIdx.x & 31;
  const int wave = threadIdx.x >> 5;
  const int o_tile = blockIdx.z * 128 + wave * 16;
  const int m  = lane & 15;   // A-matrix row for this lane
  const int kh = lane >> 4;   // 0: K {0..7,16..23}; 1: K {8..15,24..31}
  const int o = o_tile + m;
  const float* wrow = W + (size_t)o * FIN;

  v8f acc = {};
  for (int c = 0; c < FIN / 32; ++c) {
    const int i0 = c * 32;
    const int baseA0 = i0 + kh * 8;        // K block 0..15 portion
    const int baseA1 = i0 + 16 + kh * 8;   // K block 16..31 portion
    __builtin_prefetch(wrow + i0 + 128, 0, 0); // global_prefetch_b8

    const float4 wA = *reinterpret_cast<const float4*>(wrow + baseA0);
    const float4 wB = *reinterpret_cast<const float4*>(wrow + baseA0 + 4);
    const float4 wC = *reinterpret_cast<const float4*>(wrow + baseA1);
    const float4 wD = *reinterpret_cast<const float4*>(wrow + baseA1 + 4);
    const float wv[16] = {wA.x, wA.y, wA.z, wA.w, wB.x, wB.y, wB.z, wB.w,
                          wC.x, wC.y, wC.z, wC.w, wD.x, wD.y, wD.z, wD.w};
    v16h a;
#pragma unroll
    for (int j = 0; j < 8; ++j) {
      a[j] = (wv[j] > thS[baseA0 + j]) ? (_Float16)1.0f : (_Float16)0.0f;
    }
#pragma unroll
    for (int j = 0; j < 8; ++j) {
      a[8 + j] = (wv[8 + j] > thS[baseA1 + j]) ? (_Float16)1.0f : (_Float16)0.0f;
    }
    // B: lanes 0-15 hold rows K {0..7,16..23}, lanes 16-31 rows K {8..15,24..31}
    // (all columns identical, so D is column-invariant).
    v16h bb;
    const int baseB = i0 + kh * 8;
#pragma unroll
    for (int j = 0; j < 16; ++j) bb[j] = sS[baseB + j + (j >= 8 ? 8 : 0)];

    acc = __builtin_amdgcn_wmma_f32_16x16x32_f16(
        /*neg_a=*/false, a, /*neg_b=*/false, bb,
        /*c_mod=*/(short)0, acc, /*reuse_a=*/false, /*reuse_b=*/false);
  }

  // D layout: lane L holds N=L%16, M = v + 8*(L>=16). Column-0 lanes (m==0)
  // carry all 16 rows between lane 0 (M 0..7) and lane 16 (M 8..15).
  if (m == 0) {
    float* dst = yraw + tb * FOUT + o_tile + kh * 8;
#pragma unroll
    for (int j = 0; j < 8; ++j) dst[j] = acc[j];
  }
}

// ---------------------------------------------------------------------------
// K3/K5: sequential per-(b,o) scan of one layer + prep of next layer's th/s.
// Block per b; N threads (N = this layer's FOUT = next layer's FIN).
// ---------------------------------------------------------------------------
template <int N>
__global__ void scan_prep_kernel(const float* __restrict__ yraw,    // [T,B,N]
                                 const float* __restrict__ cnt0_in, // [T,B]
                                 const float* __restrict__ bvec,    // [N]
                                 const float offset,
                                 float* __restrict__ thN,           // [T,B,N]
                                 _Float16* __restrict__ sN,         // [T,B,N]
                                 float* __restrict__ cnt0_out) {    // [T,B]
  __shared__ float rng[T_STEPS];
  __shared__ int   cnt[T_STEPS];
  const int b = blockIdx.x;
  const int h = threadIdx.x; // 0..N-1
  for (int k = h; k < T_STEPS; k += N) { rng[k] = sobol_val(k); cnt[k] = 0; }
  __syncthreads();

  const float bias = bvec[h];
  float in_acc = 0.f, out_acc = 0.f, c = 0.f;
  int cum = 0;
  for (int t = 0; t < T_STEPS; ++t) {
    const size_t tb = (size_t)t * BATCH + b;
    const float y = yraw[tb * N + h] + cnt0_in[tb] +
                    ((bias > rng[t]) ? 1.0f : 0.0f) - offset;
    in_acc += y;
    const float out = (in_acc > out_acc) ? 1.0f : 0.0f;
    out_acc += out;
    // ucompare (unary ReLU): sign=(c<0); a = sign ? 1 : out; c += 2a-1
    const float a = (c < 0.0f) ? 1.0f : out;
    c += 2.0f * a - 1.0f;
    const int ai = (a > 0.5f) ? 1 : 0;
    const size_t idx = tb * N + h;
    thN[idx] = ai ? rng[cum] : rng[t - cum];
    sN[idx]  = ai ? (_Float16)1.0f : (_Float16)(-1.0f);
    int na = 1 - ai;
#pragma unroll
    for (int d = 16; d; d >>= 1) na += __shfl_down(na, d, 32);
    if ((h & 31) == 0) atomicAdd(&cnt[t], na);
    cum += ai;
  }
  __syncthreads();
  for (int k = h; k < T_STEPS; k += N)
    cnt0_out[(size_t)k * BATCH + b] = (float)cnt[k];
}

// ---------------------------------------------------------------------------
// K6a: layer 3 compare-count (fout=10 too small for WMMA tiles).
// Grid (T,B), block 320 = 10 waves; wave w reduces output o=w over 256 inputs.
// ---------------------------------------------------------------------------
__global__ void linear3_kernel(const float* __restrict__ W3,      // [10,256]
                               const float* __restrict__ th3,     // [T,B,256]
                               const _Float16* __restrict__ s3,   // [T,B,256]
                               float* __restrict__ yraw3) {       // [T,B,10]
  __shared__ float    thS[H2];
  __shared__ _Float16 sS[H2];
  const int t = blockIdx.x, b = blockIdx.y;
  const size_t tb = (size_t)t * BATCH + b;
  const int tid = threadIdx.x;
  if (tid < H2) {
    thS[tid] = th3[tb * H2 + tid];
    sS[tid]  = s3[tb * H2 + tid];
  }
  __syncthreads();
  const int o = tid >> 5, lane = tid & 31;
  const float* wrow = W3 + (size_t)o * H2;
  float acc = 0.f;
  for (int i = lane; i < H2; i += 32) {
    acc += (wrow[i] > thS[i]) ? (float)sS[i] : 0.0f;
  }
#pragma unroll
  for (int d = 16; d; d >>= 1) acc += __shfl_down(acc, d, 32);
  if (lane == 0) yraw3[tb * NOUT + o] = acc;
}

// ---------------------------------------------------------------------------
// K6b: layer-3 scan -> v = 2*y3 - 1 - prediction.  One block, 160 threads.
// ---------------------------------------------------------------------------
__global__ void scan3_kernel(const float* __restrict__ yraw3,
                             const float* __restrict__ cnt0_3,
                             const float* __restrict__ b3,
                             const float* __restrict__ pred, // [B,10]
                             float* __restrict__ v3) {       // [T,B,10]
  __shared__ float rng[T_STEPS];
  const int tid = threadIdx.x; // 0..159
  for (int k = tid; k < T_STEPS; k += 160) rng[k] = sobol_val(k);
  __syncthreads();
  const int b = tid / NOUT, o = tid % NOUT;
  const float bias = b3[o];
  const float p = pred[b * NOUT + o];
  float in_acc = 0.f, out_acc = 0.f;
  for (int t = 0; t < T_STEPS; ++t) {
    const size_t tb = (size_t)t * BATCH + b;
    const float y = yraw3[tb * NOUT + o] + cnt0_3[tb] +
                    ((bias > rng[t]) ? 1.0f : 0.0f) - 127.5f;
    in_acc += y;
    const float y3 = (in_acc > out_acc) ? 1.0f : 0.0f;
    out_acc += y3;
    v3[tb * NOUT + o] = 2.0f * y3 - 1.0f - p;
  }
}

// ---------------------------------------------------------------------------
// K7: log_softmax over the 10 outputs per (t,b).
// ---------------------------------------------------------------------------
__global__ void softmax_kernel(const float* __restrict__ v3,
                               float* __restrict__ out) {
  const int idx = blockIdx.x * blockDim.x + threadIdx.x; // (t,b) flat
  if (idx >= T_STEPS * BATCH) return;
  const float* v = v3 + (size_t)idx * NOUT;
  float m = v[0];
#pragma unroll
  for (int o = 1; o < NOUT; ++o) m = fmaxf(m, v[o]);
  float s = 0.f;
#pragma unroll
  for (int o = 0; o < NOUT; ++o) s += expf(v[o] - m);
  const float lse = m + logf(s);
  float* dst = out + (size_t)idx * NOUT;
#pragma unroll
  for (int o = 0; o < NOUT; ++o) dst[o] = v[o] - lse;
}

// ---------------------------------------------------------------------------
extern "C" void kernel_launch(void* const* d_in, const int* in_sizes, int n_in,
                              void* d_out, int out_size, void* d_ws,
                              size_t ws_size, hipStream_t stream) {
  const float* w1   = (const float*)d_in[0];
  const float* b1   = (const float*)d_in[1];
  const float* w2   = (const float*)d_in[2];
  const float* b2   = (const float*)d_in[3];
  const float* w3   = (const float*)d_in[4];
  const float* b3   = (const float*)d_in[5];
  const float* pred = (const float*)d_in[6];
  const int*   x    = (const int*)d_in[7];

  char* ws = (char*)d_ws;
  auto alloc = [&](size_t bytes) -> char* {
    char* p = ws;
    ws += (bytes + 255) & ~(size_t)255;
    return p;
  };
  const size_t TB = (size_t)T_STEPS * BATCH;
  float*    th1   = (float*)   alloc(TB * IN1 * 4);
  _Float16* s1    = (_Float16*)alloc(TB * IN1 * 2);
  float*    yraw1 = (float*)   alloc(TB * H1 * 4);
  float*    c0_1  = (float*)   alloc(TB * 4);
  float*    th2   = (float*)   alloc(TB * H1 * 4);
  _Float16* s2    = (_Float16*)alloc(TB * H1 * 2);
  float*    yraw2 = (float*)   alloc(TB * H2 * 4);
  float*    c0_2  = (float*)   alloc(TB * 4);
  float*    th3   = (float*)   alloc(TB * H2 * 4);
  _Float16* s3    = (_Float16*)alloc(TB * H2 * 2);
  float*    yraw3 = (float*)   alloc(TB * NOUT * 4);
  float*    c0_3  = (float*)   alloc(TB * 4);
  float*    v3    = (float*)   alloc(TB * NOUT * 4);

  prep1_kernel<<<BATCH, IN1, 0, stream>>>(x, th1, s1, c0_1);

  ulinear_wmma_kernel<IN1, H1>
      <<<dim3(T_STEPS, BATCH, H1 / 128), 256, 0, stream>>>(w1, th1, s1, yraw1);

  scan_prep_kernel<H1><<<BATCH, H1, 0, stream>>>(yraw1, c0_1, b1,
                                                 (IN1 - 1) * 0.5f, th2, s2,
                                                 c0_2);

  ulinear_wmma_kernel<H1, H2>
      <<<dim3(T_STEPS, BATCH, H2 / 128), 256, 0, stream>>>(w2, th2, s2, yraw2);

  scan_prep_kernel<H2><<<BATCH, H2, 0, stream>>>(yraw2, c0_2, b2,
                                                 (H1 - 1) * 0.5f, th3, s3,
                                                 c0_3);

  linear3_kernel<<<dim3(T_STEPS, BATCH), 320, 0, stream>>>(w3, th3, s3, yraw3);

  scan3_kernel<<<1, BATCH * NOUT, 0, stream>>>(yraw3, c0_3, b3, pred, v3);

  softmax_kernel<<<(T_STEPS * BATCH + 255) / 256, 256, 0, stream>>>(
      v3, (float*)d_out);
}
